// TransitionGNN_38233798869138
// MI455X (gfx1250) — matrix operand order repaired and checked by reading, not compiled, not run on hardware
//
#include <hip/hip_runtime.h>

typedef __attribute__((ext_vector_type(16))) _Float16 v16h;
typedef __attribute__((ext_vector_type(8)))  float    v8f;

#define KN 32   // nodes per graph
#define NH 16   // hidden width

union HV {
    v16h v;
    unsigned int d[8];
    _Float16 h[16];
};

__device__ __forceinline__ float fast_tanh(float x) {
    // tanh(x) = 1 - 2/(e^{2x}+1); saturates correctly at +-1
    float e = __expf(2.f * x);
    return 1.f - 2.f * __builtin_amdgcn_rcpf(e + 1.f);
}
__device__ __forceinline__ float fast_sigmoid(float x) {
    return __builtin_amdgcn_rcpf(1.f + __expf(-x));
}

// Build B-matrix (32x16 f16, K=16 real + 16 zero-pad) for v_wmma_f32_16x16x32_f16.
// On entry each lane holds 8 f32 = features [8*(lane/16) .. +8) of edge (lane&15).
// B layout: lane<16 -> halves 0..7 = features 0..7 of edge lane (K=0..7);
//           lane>=16 -> halves 0..7 = features 8..15 of edge lane-16 (K=8..15); halves 8..15 = zeros.
__device__ __forceinline__ v16h make_b(const float* vals8, int lane) {
    HV b;
#pragma unroll
    for (int r = 0; r < 8; ++r) b.h[r] = (_Float16)vals8[r];
#pragma unroll
    for (int r = 4; r < 8; ++r) b.d[r] = 0u;
    unsigned pd[4];
#pragma unroll
    for (int r = 0; r < 4; ++r)   // SWAPX16: xor=0x10, and=0x1f
        pd[r] = (unsigned)__builtin_amdgcn_ds_swizzle((int)b.d[r], 0x401F);
    const bool lo = lane < 16;
#pragma unroll
    for (int r = 0; r < 4; ++r) {
        unsigned own = b.d[r];
        b.d[r]     = own;
        b.d[r + 4] = 0u;
    }
    return b.v;
}

// Build A-matrix = W^T (16 x K16, zero-padded to K=32).
// 16-bit A layout: lane<16: M=lane, halves0..7 = K 0..7 ; lane>=16: M=lane-16, halves0..7 = K 8..15.
// W is row-major (fan_in x fan_out): W^T[m,k] = W[k*fan_out + m].
__device__ __forceinline__ v16h load_wT(const float* __restrict__ W, int fan_out,
                                        int valid_out, int lane) {
    HV a;
    const int m  = lane & 15;
    const int kb = (lane >> 4) * 8;
#pragma unroll
    for (int h = 0; h < 8; ++h)
        a.h[h] = (m < valid_out) ? (_Float16)W[(kb + h) * fan_out + m] : (_Float16)0.f;
#pragma unroll
    for (int r = 4; r < 8; ++r) a.d[r] = 0u;
    return a.v;
}

// Bias in C/D layout: lane's VGPR r = out-feature (8*(lane/16)+r), broadcast over N.
__device__ __forceinline__ v8f load_bias(const float* __restrict__ b, int valid, int lane) {
    v8f c = {};
    const int off = (lane >> 4) * 8;
#pragma unroll
    for (int r = 0; r < 8; ++r) {
        int f = off + r;
        c[r] = (f < valid) ? b[f] : 0.f;
    }
    return c;
}

__global__ __launch_bounds__(256) void gnn_fused_kernel(
    const float* __restrict__ ef,  const float* __restrict__ action,
    const float* __restrict__ Wei, const float* __restrict__ bei,
    const float* __restrict__ Wn,  const float* __restrict__ bn,
    const float* __restrict__ We,  const float* __restrict__ be,
    const float* __restrict__ Wf1, const float* __restrict__ bf1,
    const float* __restrict__ Wf2, const float* __restrict__ bf2,
    const float* __restrict__ Wf3, const float* __restrict__ bf3,
    float* __restrict__ out)
{
    __shared__ float part[256 * 8];   // per-thread partial feature sums
    __shared__ float sarr[KN * 8];    // s[i,0..6], slot 7 = 32*action
    __shared__ float hnsum[KN * NH];
    __shared__ float hnod[KN * NH];
    __shared__ float Pp[KN * NH];     // hn@We_top + be
    __shared__ float Qq[KN * NH];     // hn@We_bot
    __shared__ __align__(16) float ostage[8 * 120];  // per-wave output staging (112 used)

    const int n    = blockIdx.x;
    const int tid  = threadIdx.x;
    const int lane = tid & 31;

    const float* efg = ef + (size_t)n * (KN * KN * 7);

    // ---- Stage 1: s[i,f] = sum_j ef[n,i,j,f] ----
    // Each thread: 28 contiguous floats (112B, 16B-aligned) -> 7x b128 loads.
    {
        const int i = tid >> 3, g = tid & 7;
        const float4* p4 = (const float4*)(efg + (i * KN + g * 4) * 7);
        float buf[28];
#pragma unroll
        for (int q = 0; q < 7; ++q) ((float4*)buf)[q] = p4[q];
        float acc[7];
#pragma unroll
        for (int f = 0; f < 7; ++f) acc[f] = 0.f;
#pragma unroll
        for (int e = 0; e < 28; ++e) acc[e % 7] += buf[e];
#pragma unroll
        for (int f = 0; f < 7; ++f) part[tid * 8 + f] = acc[f];
    }
    __syncthreads();
    if (tid < 224) {
        const int i = tid / 7, f = tid % 7;
        float s = 0.f;
#pragma unroll
        for (int g = 0; g < 8; ++g) s += part[(i * 8 + g) * 8 + f];
        sarr[i * 8 + f] = s;
    }
    if (tid < KN) sarr[tid * 8 + 7] = 32.f * action[n];
    __syncthreads();

    // ---- Stage 2: node pipeline (tiny, scalar f32) ----
    // hnsum[i] = s_ext[i] @ Wei + 32*bei   (sum_j of embed, by linearity)
    for (int idx = tid; idx < KN * NH; idx += 256) {
        const int i = idx >> 4, h = idx & 15;
        float acc = 32.f * bei[h];
#pragma unroll
        for (int k = 0; k < 8; ++k) acc += sarr[i * 8 + k] * Wei[k * NH + h];
        hnsum[idx] = acc;
    }
    __syncthreads();
    // hn[i] = tanh(hnsum[i] @ Wn + bn)
    for (int idx = tid; idx < KN * NH; idx += 256) {
        const int i = idx >> 4, h = idx & 15;
        float acc = bn[h];
#pragma unroll
        for (int k = 0; k < 16; ++k) acc += hnsum[i * 16 + k] * Wn[k * NH + h];
        hnod[idx] = fast_tanh(acc);
    }
    __syncthreads();
    // P[i] = hn[i]@We_top + be ; Q[j] = hn[j]@We_bot
    for (int idx = tid; idx < KN * NH; idx += 256) {
        const int i = idx >> 4, h = idx & 15;
        float pa = be[h], qa = 0.f;
#pragma unroll
        for (int k = 0; k < 16; ++k) {
            const float hv = hnod[i * 16 + k];
            pa += hv * We[k * NH + h];
            qa += hv * We[(16 + k) * NH + h];
        }
        Pp[idx] = pa;
        Qq[idx] = qa;
    }
    __syncthreads();

    // ---- Stage 3: per-edge MLP via WMMA (weights in A, activations in B) ----
    const v16h a1 = load_wT(Wf1, 16, 16, lane);
    const v16h a2 = load_wT(Wf2, 16, 16, lane);
    const v16h a3 = load_wT(Wf3, 7, 7, lane);
    const v8f  c1 = load_bias(bf1, 16, lane);
    const v8f  c2 = load_bias(bf2, 16, lane);
    const v8f  c3 = load_bias(bf3, 7, lane);

    const int wave = tid >> 5;
    float* stg = ostage + wave * 120;
    float* outg = out + (size_t)n * (KN * KN * 7);

    for (int t = wave; t < 64; t += 8) {           // 64 tiles of 16 edges
        const int e  = lane & 15;
        const int i  = t >> 1;                     // edge E = 16t+e ; i = E/32
        const int j  = ((t & 1) << 4) | e;         // j = E%32
        const int fb = (lane >> 4) * 8;

        // he2 = tanh(P[i] + Q[j] + be)  (be folded into P)
        float v0[8];
#pragma unroll
        for (int h = 0; h < 8; ++h)
            v0[h] = fast_tanh(Pp[i * 16 + fb + h] + Qq[j * 16 + fb + h]);
        v16h b1 = make_b(v0, lane);
        v8f d1 = __builtin_amdgcn_wmma_f32_16x16x32_f16(
            false, a1, false, b1, (short)0, c1, false, false);

        float v1[8];
#pragma unroll
        for (int h = 0; h < 8; ++h) v1[h] = d1[h] > 0.f ? d1[h] : 0.f;
        v16h b2 = make_b(v1, lane);
        v8f d2 = __builtin_amdgcn_wmma_f32_16x16x32_f16(
            false, a2, false, b2, (short)0, c2, false, false);

        float v2[8];
#pragma unroll
        for (int h = 0; h < 8; ++h) v2[h] = d2[h] > 0.f ? d2[h] : 0.f;
        v16h b3 = make_b(v2, lane);
        v8f d3 = __builtin_amdgcn_wmma_f32_16x16x32_f16(
            false, a3, false, b3, (short)0, c3, false, false);

        // out = sigmoid(tanh(d3)); lane<16 holds out-features 0..7 of edge e.
        // Stage through LDS, then 28 lanes emit coalesced b128 stores
        // (tile region = 448B contiguous, 16B-aligned).
        if (lane < 16) {
#pragma unroll
            for (int r = 0; r < 7; ++r)
                stg[e * 7 + r] = fast_sigmoid(fast_tanh(d3[r]));
        }
        asm volatile("s_wait_dscnt 0x0" ::: "memory");   // intra-wave LDS ordering
        if (lane < 28) {
            float4 vv = ((const float4*)stg)[lane];
            *(float4*)(outg + (size_t)t * 112 + lane * 4) = vv;
        }
    }
}

extern "C" void kernel_launch(void* const* d_in, const int* in_sizes, int n_in,
                              void* d_out, int out_size, void* d_ws, size_t ws_size,
                              hipStream_t stream) {
    const float* ef     = (const float*)d_in[0];
    const float* action = (const float*)d_in[1];
    const float* Wei    = (const float*)d_in[2];
    const float* bei    = (const float*)d_in[3];
    const float* Wn     = (const float*)d_in[4];
    const float* bn     = (const float*)d_in[5];
    const float* We     = (const float*)d_in[6];
    const float* be     = (const float*)d_in[7];
    const float* Wf1    = (const float*)d_in[8];
    const float* bf1    = (const float*)d_in[9];
    const float* Wf2    = (const float*)d_in[10];
    const float* bf2    = (const float*)d_in[11];
    const float* Wf3    = (const float*)d_in[12];
    const float* bf3    = (const float*)d_in[13];
    float* out = (float*)d_out;

    const int N = in_sizes[0] / (KN * KN * 7);     // 4096 graphs
    gnn_fused_kernel<<<dim3(N), dim3(256), 0, stream>>>(
        ef, action, Wei, bei, Wn, bn, We, be,
        Wf1, bf1, Wf2, bf2, Wf3, bf3, out);
}
